// GCNMultiClass_46119358825091
// MI455X (gfx1250) — matrix operand order repaired and checked by reading, not compiled, not run on hardware
//
#include <hip/hip_runtime.h>

// ---------------- problem constants ----------------
#define NN    50000
#define EE    500000
#define GG    512
#define INC   6
#define HIDC  32
#define EXTRA 4
#define EPSV  1e-5f
#define NTILES (EE/16)   // 31250 exact

typedef __attribute__((ext_vector_type(16))) __bf16 v16bf;
typedef __attribute__((ext_vector_type(8)))  float  v8f;

union V16 { v16bf v; unsigned int u[8]; unsigned short h[16]; };
union V8F { v8f v; float f[8]; };

__device__ __forceinline__ unsigned short f2bf(float f) {
  unsigned int u = __float_as_uint(f);
  unsigned int r = u + 0x7FFFu + ((u >> 16) & 1u);   // round-to-nearest-even
  return (unsigned short)(r >> 16);
}

// packed f32x2 -> bf16x2 single-instruction convert (gfx1250)
__device__ __forceinline__ unsigned int cvt_pk_bf16(float lo, float hi) {
  unsigned int d;
  asm("v_cvt_pk_bf16_f32 %0, %1, %2" : "=v"(d) : "v"(lo), "v"(hi));
  return d;
}

__device__ __forceinline__ unsigned int pkmul_bf16(unsigned int a, unsigned int b) {
  unsigned int d;
  asm volatile("v_pk_mul_bf16 %0, %1, %2" : "=v"(d) : "v"(a), "v"(b));
  return d;
}

__device__ __forceinline__ void wave_lds_fence() {
  asm volatile("s_wait_dscnt 0x0" ::: "memory");
}

__device__ __forceinline__ void wait_async0() {
  asm volatile("s_wait_asynccnt 0x0" ::: "memory");
}

// low 32 bits of a generic LDS pointer == wave-relative LDS byte offset
__device__ __forceinline__ unsigned int lds_off(const void* p) {
  return (unsigned int)(unsigned long long)p;
}

// async memory -> LDS copy, 16B per lane, ASYNCcnt-tracked (gfx1250)
__device__ __forceinline__ void async_b128(unsigned int ldsAddr, const void* base,
                                           unsigned int voff) {
  asm volatile("global_load_async_to_lds_b128 %0, %1, %2 offset:0"
               :: "v"(ldsAddr), "v"(voff), "s"(base) : "memory");
}

__device__ __forceinline__ v8f wmma_bf16(v16bf a, v16bf b, v8f c) {
  return __builtin_amdgcn_wmma_f32_16x16x32_bf16(false, a, false, b, (short)0, c, false, false);
}

__device__ __forceinline__ v16bf ldsB(const unsigned short* p) {
  V16 r;
  const unsigned int* q = (const unsigned int*)p;   // 32B-aligned fragments
#pragma unroll
  for (int j = 0; j < 8; ++j) r.u[j] = q[j];
  return r.v;
}

// ---------------- weight panel packing ----------------
// B fragment layout per lane L (wave32): column n = L&15 of the 16-wide n-tile,
// K = (L>>4)*16 + kk, kk = element index 0..15 of the v16bf.
// Fragment for (step s, ntile t) stored at out[((s*2+t)*32 + L)*16 + kk].

__global__ void packMLP(const float* __restrict__ w1, unsigned short* __restrict__ out) {
  int idx = blockIdx.x * blockDim.x + threadIdx.x;
  if (idx >= 1024) return;
  int t = idx >> 9, r = idx & 511, L = r >> 4, kk = r & 15;
  int K = (L >> 4) * 16 + kk, o = t * 16 + (L & 15);
  float v = (K < 5) ? w1[K * 32 + o] : 0.f;
  out[(t * 32 + L) * 16 + kk] = f2bf(v);
}

// Layer1 reshaped panel: K2'=288.  c = k*8+i (i<8, x padded), bias rows at c>=256.
__global__ void packW1(const float* __restrict__ w2, const float* __restrict__ b2,
                       unsigned short* __restrict__ out) {
  int idx = blockIdx.x * blockDim.x + threadIdx.x;
  if (idx >= 9 * 1024) return;
  int s = idx >> 10, r = idx & 1023, t = r >> 9, q = r & 511, L = q >> 4, kk = q & 15;
  int K = (L >> 4) * 16 + kk, o = t * 16 + (L & 15);
  int c = s * 32 + K;
  float v = 0.f;
  if (c < 256) { int k = c >> 3, i = c & 7; if (i < 6) v = w2[k * 192 + i * 32 + o]; }
  else         { int j = c - 256;           if (j < 6) v = b2[j * 32 + o]; }
  out[((s * 2 + t) * 32 + L) * 16 + kk] = f2bf(v);
}

// Layer2 reshaped panel: K2'=1056.  c = k*32+i, bias rows at c>=1024.
__global__ void packW2(const float* __restrict__ w2, const float* __restrict__ b2,
                       unsigned short* __restrict__ out) {
  int idx = blockIdx.x * blockDim.x + threadIdx.x;
  if (idx >= 33 * 1024) return;
  int s = idx >> 10, r = idx & 1023, t = r >> 9, q = r & 511, L = q >> 4, kk = q & 15;
  int K = (L >> 4) * 16 + kk, o = t * 16 + (L & 15);
  int c = s * 32 + K;
  float v;
  if (c < 1024) { int k = c >> 5, i = c & 31; v = w2[k * 1024 + i * 32 + o]; }
  else          { int j = c - 1024;           v = b2[j * 32 + o]; }
  out[((s * 2 + t) * 32 + L) * 16 + kk] = f2bf(v);
}

__global__ void prepXpad(const float* __restrict__ x, unsigned short* __restrict__ xpad) {
  int idx = blockIdx.x * blockDim.x + threadIdx.x;
  if (idx >= NN * 8) return;
  int n = idx >> 3, i = idx & 7;
  xpad[idx] = f2bf(i < 6 ? x[n * 6 + i] : 0.f);
}

__global__ void countK(const int* __restrict__ dst, float* __restrict__ cnt) {
  int e = blockIdx.x * blockDim.x + threadIdx.x;
  if (e >= EE) return;
  atomicAdd(&cnt[dst[e]], 1.0f);
}

// ---------------- fused NNConv edge kernel ----------------
// One wave per 16-edge tile:
//   h = relu(edge_attr @ w1 + b1)       : 2x WMMA (K padded 5->32), staged in LDS
//   msg = [Y | xs] @ [Wr ; Br]          : S k-steps x 2 n-tiles of WMMA,
//                                         Y rows built per-lane with v_pk_mul_bf16
//   scatter-mean numerator via global fp32 atomics
// B panels land in LDS via global_load_async_to_lds_b128 (ASYNCcnt).
template <int LAYER>
__global__ __launch_bounds__(256) void edgeK(
    const int* __restrict__ srcI, const int* __restrict__ dstI,
    const float* __restrict__ eattr,
    const unsigned short* __restrict__ xsrc,      // LAYER==1: xpad[N,8]; LAYER==2: h1b[N,32]
    const unsigned short* __restrict__ bhPack,    // 1024 bf16
    const unsigned short* __restrict__ bwPack,    // S*1024 bf16
    const float* __restrict__ b1,                 // 32 (edge-MLP hidden bias)
    float* __restrict__ agg)
{
  constexpr int S = (LAYER == 1) ? 9 : 33;
  __shared__ __align__(16) unsigned short sBh[1024];
  __shared__ __align__(16) unsigned short sBW[S * 1024];
  __shared__ __align__(16) unsigned short sH[8][16 * 32];
  __shared__ int sDst[8][16];

  // ---- async preload of the packed B panels into LDS ----
  {
    unsigned int ldsH = lds_off(&sBh[0]);
    unsigned int ldsW = lds_off(&sBW[0]);
    for (unsigned int b = threadIdx.x * 16u; b < 2048u; b += blockDim.x * 16u)
      async_b128(ldsH + b, bhPack, b);
    for (unsigned int b = threadIdx.x * 16u; b < (unsigned)S * 2048u; b += blockDim.x * 16u)
      async_b128(ldsW + b, bwPack, b);
    wait_async0();
    __syncthreads();
  }

  const int wave = threadIdx.x >> 5;
  const int lane = threadIdx.x & 31;
  const int lm   = lane & 15;         // A row index (edge within tile), also B/C column n
  const int g    = lane >> 4;         // lane group
  const float bia0 = b1[lm];
  const float bia1 = b1[16 + lm];

  const int wgid     = blockIdx.x * (blockDim.x >> 5) + wave;
  const int totWaves = gridDim.x * (blockDim.x >> 5);

  V8F zf;
#pragma unroll
  for (int j = 0; j < 8; ++j) zf.f[j] = 0.f;

  for (int tile = wgid; tile < NTILES; tile += totWaves) {
    const int e    = tile * 16 + lm;
    const int sidx = srcI[e];
    const int didx = dstI[e];

    // ---- edge-MLP hidden layer: A = edge_attr row (K padded to 32) ----
    V16 a1;
#pragma unroll
    for (int j = 0; j < 8; ++j) a1.u[j] = 0u;
    if (g == 0) {
      const float* ea = eattr + (size_t)e * 5;
      a1.u[0] = cvt_pk_bf16(ea[0], ea[1]);
      a1.u[1] = cvt_pk_bf16(ea[2], ea[3]);
      a1.u[2] = cvt_pk_bf16(ea[4], 0.f);
    }
    v8f c0 = wmma_bf16(a1.v, ldsB(&sBh[(0 * 32 + lane) * 16]), zf.v);
    v8f c1 = wmma_bf16(a1.v, ldsB(&sBh[(1 * 32 + lane) * 16]), zf.v);

    // stage h (bf16, post bias+relu) and dst into wave-local LDS
    V8F cc0, cc1; cc0.v = c0; cc1.v = c1;
#pragma unroll
    for (int r = 0; r < 8; ++r) {
      int row = g * 8 + r;
      float v0 = fmaxf(cc0.f[r] + bia0, 0.f);     // column lm
      float v1 = fmaxf(cc1.f[r] + bia1, 0.f);     // column lm+16
      unsigned int pk = cvt_pk_bf16(v0, v1);
      unsigned short* p = &sH[wave][row * 32 + lm];
      p[0]  = (unsigned short)pk;                  // ds_store_b16
      p[16] = (unsigned short)(pk >> 16);          // ds_store_b16_d16_hi
    }
    if (g == 0) sDst[wave][lm] = didx;
    wave_lds_fence();

    // ---- xs fragment for this lane (invariant across all K-steps) ----
    unsigned int xp[8];
    if (LAYER == 1) {
      const unsigned int* q = (const unsigned int*)(xsrc + (size_t)sidx * 8);
#pragma unroll
      for (int j = 0; j < 4; ++j) { xp[j] = q[j]; xp[4 + j] = q[j]; }
    } else {
      const unsigned int* q0 = (const unsigned int*)(xsrc + (size_t)sidx * 32 + g * 16);
      const unsigned int* q1 = (const unsigned int*)(xsrc + (size_t)sidx * 32 + g * 16 + 8);
#pragma unroll
      for (int j = 0; j < 4; ++j) { xp[j] = q0[j]; xp[4 + j] = q1[j]; }
    }

    // ---- msg = Y @ packed panel (S-1 scaled steps) ----
    v8f d0 = zf.v, d1 = zf.v;
#pragma unroll 4
    for (int s = 0; s < S - 1; ++s) {
      V16 av;
      if (LAYER == 2) {
        unsigned short hs = sH[wave][lm * 32 + s];
        unsigned int hp = (unsigned int)hs | ((unsigned int)hs << 16);
#pragma unroll
        for (int j = 0; j < 8; ++j) av.u[j] = pkmul_bf16(hp, xp[j]);
      } else {
        unsigned short ha = sH[wave][lm * 32 + 4 * s + 2 * g];
        unsigned short hb = sH[wave][lm * 32 + 4 * s + 2 * g + 1];
        unsigned int hpa = (unsigned int)ha | ((unsigned int)ha << 16);
        unsigned int hpb = (unsigned int)hb | ((unsigned int)hb << 16);
#pragma unroll
        for (int j = 0; j < 4; ++j) av.u[j]     = pkmul_bf16(hpa, xp[j]);
#pragma unroll
        for (int j = 0; j < 4; ++j) av.u[4 + j] = pkmul_bf16(hpb, xp[j]);
      }
      d0 = wmma_bf16(av.v, ldsB(&sBW[((s * 2 + 0) * 32 + lane) * 16]), d0);
      d1 = wmma_bf16(av.v, ldsB(&sBW[((s * 2 + 1) * 32 + lane) * 16]), d1);
    }
    // peeled bias block: Y = xs raw
    {
      V16 av;
#pragma unroll
      for (int j = 0; j < 8; ++j) av.u[j] = xp[j];
      d0 = wmma_bf16(av.v, ldsB(&sBW[(((S - 1) * 2 + 0) * 32 + lane) * 16]), d0);
      d1 = wmma_bf16(av.v, ldsB(&sBW[(((S - 1) * 2 + 1) * 32 + lane) * 16]), d1);
    }

    // ---- scatter-mean numerator ----
    int dr[8];
#pragma unroll
    for (int r = 0; r < 8; ++r) dr[r] = sDst[wave][g * 8 + r];
    V8F dd0, dd1; dd0.v = d0; dd1.v = d1;
#pragma unroll
    for (int r = 0; r < 8; ++r) {
      atomicAdd(&agg[(size_t)dr[r] * 32 + lm],      dd0.f[r]);
      atomicAdd(&agg[(size_t)dr[r] * 32 + 16 + lm], dd1.f[r]);
    }
  }
}

// ---------------- node-side kernels ----------------
template <int IN>
__global__ void nodeFinish(const float* __restrict__ agg, const float* __restrict__ cnt,
                           const float* __restrict__ xin, const float* __restrict__ root,
                           const float* __restrict__ bias, float* __restrict__ pre,
                           float* __restrict__ sum, float* __restrict__ sq)
{
  __shared__ float ss[32], sv[32];
  int tid = blockIdx.x * blockDim.x + threadIdx.x;
  if (threadIdx.x < 32) { ss[threadIdx.x] = 0.f; sv[threadIdx.x] = 0.f; }
  __syncthreads();
  float v = 0.f; int c = 0;
  if (tid < NN * 32) {
    int n = tid >> 5; c = tid & 31;
    float a = agg[tid] / fmaxf(cnt[n], 1.0f);
    float rs = 0.f;
#pragma unroll
    for (int i = 0; i < IN; ++i) rs += xin[n * IN + i] * root[i * 32 + c];
    v = a + rs + bias[c];
    pre[tid] = v;
  }
  atomicAdd(&ss[c], v);
  atomicAdd(&sv[c], v * v);
  __syncthreads();
  if (threadIdx.x < 32) {
    atomicAdd(&sum[threadIdx.x], ss[threadIdx.x]);
    atomicAdd(&sq[threadIdx.x],  sv[threadIdx.x]);
  }
}

__global__ void bnApply1(const float* __restrict__ pre, const float* __restrict__ sum,
                         const float* __restrict__ sq, const float* __restrict__ gam,
                         const float* __restrict__ bet, float* __restrict__ h1,
                         unsigned short* __restrict__ h1b)
{
  int tid = blockIdx.x * blockDim.x + threadIdx.x;
  if (tid >= NN * 32) return;
  int c = tid & 31;
  float mu  = sum[c] * (1.0f / NN);
  float var = sq[c]  * (1.0f / NN) - mu * mu;
  float v = (pre[tid] - mu) * rsqrtf(var + EPSV) * gam[c] + bet[c];
  v = fmaxf(v, 0.f);
  h1[tid] = v;
  h1b[tid] = f2bf(v);
}

__global__ void bnApply2Pool(const float* __restrict__ pre, const float* __restrict__ sum,
                             const float* __restrict__ sq, const float* __restrict__ gam,
                             const float* __restrict__ bet, const int* __restrict__ batch,
                             float* __restrict__ psum)
{
  int tid = blockIdx.x * blockDim.x + threadIdx.x;
  if (tid >= NN * 32) return;
  int n = tid >> 5, c = tid & 31;
  float mu  = sum[c] * (1.0f / NN);
  float var = sq[c]  * (1.0f / NN) - mu * mu;
  float v = (pre[tid] - mu) * rsqrtf(var + EPSV) * gam[c] + bet[c];
  v = fmaxf(v, 0.f);
  atomicAdd(&psum[(size_t)batch[n] * 32 + c], v);
}

__global__ void pcntK(const int* __restrict__ batch, float* __restrict__ pcnt) {
  int n = blockIdx.x * blockDim.x + threadIdx.x;
  if (n >= NN) return;
  atomicAdd(&pcnt[batch[n]], 1.0f);
}

__global__ void finalK(const float* __restrict__ psum, const float* __restrict__ pcnt,
                       const float* __restrict__ gattr, const float* __restrict__ fcw,
                       const float* __restrict__ fcb, float* __restrict__ out)
{
  int tid = blockIdx.x * blockDim.x + threadIdx.x;
  if (tid >= GG * 3) return;
  int gI = tid / 3, j = tid % 3;
  float ic = 1.0f / fmaxf(pcnt[gI], 1.0f);
  float acc = fcb[j];
#pragma unroll
  for (int c = 0; c < 32; ++c) acc += psum[gI * 32 + c] * ic * fcw[c * 3 + j];
#pragma unroll
  for (int k = 0; k < 4; ++k)  acc += gattr[gI * 4 + k] * fcw[(32 + k) * 3 + j];
  out[tid] = acc;
}

// ---------------- workspace layout (bytes) ----------------
#define OFF_AGG1   ((size_t)0)
#define OFF_AGG2   (OFF_AGG1 + (size_t)NN * 32 * 4)
#define OFF_CNT    (OFF_AGG2 + (size_t)NN * 32 * 4)
#define OFF_STATS  (OFF_CNT  + (size_t)NN * 4)          // sum1,sq1,sum2,sq2 (4*32 f32)
#define OFF_PSUM   (OFF_STATS + (size_t)4 * 32 * 4)
#define OFF_PCNT   (OFF_PSUM + (size_t)GG * 32 * 4)
#define ZERO_BYTES (OFF_PCNT + (size_t)GG * 4)
#define OFF_H1PRE  (ZERO_BYTES)
#define OFF_H1     (OFF_H1PRE + (size_t)NN * 32 * 4)
#define OFF_H2PRE  (OFF_H1    + (size_t)NN * 32 * 4)
#define OFF_H1B    (OFF_H2PRE + (size_t)NN * 32 * 4)
#define OFF_XPAD   (OFF_H1B   + (size_t)NN * 32 * 2)
#define OFF_BH1    (OFF_XPAD  + (size_t)NN * 8 * 2)
#define OFF_BH2    (OFF_BH1   + (size_t)1024 * 2)
#define OFF_BW1    (OFF_BH2   + (size_t)1024 * 2)
#define OFF_BW2    (OFF_BW1   + (size_t)9 * 1024 * 2)
#define WS_TOTAL   (OFF_BW2   + (size_t)33 * 1024 * 2)

extern "C" void kernel_launch(void* const* d_in, const int* in_sizes, int n_in,
                              void* d_out, int out_size, void* d_ws, size_t ws_size,
                              hipStream_t stream) {
  (void)in_sizes; (void)n_in; (void)out_size;
  if (ws_size < WS_TOTAL) return;

  const float* x      = (const float*)d_in[0];
  const int*   eidx   = (const int*)d_in[1];
  const float* eattr  = (const float*)d_in[2];
  const int*   batch  = (const int*)d_in[3];
  const float* gattr  = (const float*)d_in[4];
  const float* e1w1   = (const float*)d_in[5];
  const float* e1b1   = (const float*)d_in[6];
  const float* e1w2   = (const float*)d_in[7];
  const float* e1b2   = (const float*)d_in[8];
  const float* root1  = (const float*)d_in[9];
  const float* bias1  = (const float*)d_in[10];
  const float* bn1g   = (const float*)d_in[11];
  const float* bn1b   = (const float*)d_in[12];
  const float* e2w1   = (const float*)d_in[13];
  const float* e2b1   = (const float*)d_in[14];
  const float* e2w2   = (const float*)d_in[15];
  const float* e2b2   = (const float*)d_in[16];
  const float* root2  = (const float*)d_in[17];
  const float* bias2  = (const float*)d_in[18];
  const float* bn2g   = (const float*)d_in[19];
  const float* bn2b   = (const float*)d_in[20];
  const float* fcw    = (const float*)d_in[21];
  const float* fcb    = (const float*)d_in[22];
  const int* srcI = eidx;
  const int* dstI = eidx + EE;

  char* ws = (char*)d_ws;
  float* agg1  = (float*)(ws + OFF_AGG1);
  float* agg2  = (float*)(ws + OFF_AGG2);
  float* cnt   = (float*)(ws + OFF_CNT);
  float* stats = (float*)(ws + OFF_STATS);  // [sum1|sq1|sum2|sq2]
  float* psum  = (float*)(ws + OFF_PSUM);
  float* pcnt  = (float*)(ws + OFF_PCNT);
  float* h1pre = (float*)(ws + OFF_H1PRE);
  float* h1    = (float*)(ws + OFF_H1);
  float* h2pre = (float*)(ws + OFF_H2PRE);
  unsigned short* h1b  = (unsigned short*)(ws + OFF_H1B);
  unsigned short* xpad = (unsigned short*)(ws + OFF_XPAD);
  unsigned short* bh1  = (unsigned short*)(ws + OFF_BH1);
  unsigned short* bh2  = (unsigned short*)(ws + OFF_BH2);
  unsigned short* bw1  = (unsigned short*)(ws + OFF_BW1);
  unsigned short* bw2  = (unsigned short*)(ws + OFF_BW2);

  hipMemsetAsync(ws, 0, ZERO_BYTES, stream);

  prepXpad<<<(NN * 8 + 255) / 256, 256, 0, stream>>>(x, xpad);
  packMLP<<<4, 256, 0, stream>>>(e1w1, bh1);
  packMLP<<<4, 256, 0, stream>>>(e2w1, bh2);
  packW1<<<36, 256, 0, stream>>>(e1w2, e1b2, bw1);
  packW2<<<132, 256, 0, stream>>>(e2w2, e2b2, bw2);
  countK<<<(EE + 255) / 256, 256, 0, stream>>>(dstI, cnt);

  // Layer 1
  edgeK<1><<<256, 256, 0, stream>>>(srcI, dstI, eattr, xpad, bh1, bw1, e1b1, agg1);
  nodeFinish<6><<<(NN * 32 + 255) / 256, 256, 0, stream>>>(agg1, cnt, x, root1, bias1,
                                                           h1pre, stats + 0, stats + 32);
  bnApply1<<<(NN * 32 + 255) / 256, 256, 0, stream>>>(h1pre, stats + 0, stats + 32,
                                                      bn1g, bn1b, h1, h1b);
  // Layer 2
  edgeK<2><<<256, 256, 0, stream>>>(srcI, dstI, eattr, h1b, bh2, bw2, e2b1, agg2);
  nodeFinish<32><<<(NN * 32 + 255) / 256, 256, 0, stream>>>(agg2, cnt, h1, root2, bias2,
                                                            h2pre, stats + 64, stats + 96);
  bnApply2Pool<<<(NN * 32 + 255) / 256, 256, 0, stream>>>(h2pre, stats + 64, stats + 96,
                                                          bn2g, bn2b, batch, psum);
  // Pool + FC
  pcntK<<<(NN + 255) / 256, 256, 0, stream>>>(batch, pcnt);
  finalK<<<(GG * 3 + 255) / 256, 256, 0, stream>>>(psum, pcnt, gattr, fcw, fcb,
                                                   (float*)d_out);
}
